// EdgeBlock_84069689852538
// MI455X (gfx1250) — matrix operand order repaired and checked by reading, not compiled, not run on hardware
//
#include <hip/hip_runtime.h>

typedef __attribute__((ext_vector_type(16))) __bf16        v16bf;
typedef __attribute__((ext_vector_type(8)))  float         v8f;
typedef __attribute__((ext_vector_type(4)))  unsigned int  u32x4;

// LDS layout (bytes). Row strides padded (+8 bf16) so 16 lanes rotate across
// all 64 banks for b128 fragment loads.
#define W1T_OFF   0          // [128 n][264 k] bf16, stride 528 B
#define W1T_STR   528
#define W2T_OFF   67584      // [64 n][136 k] bf16, stride 272 B
#define W2T_STR   272
#define B1_OFF    84992      // 128 f32
#define B2_OFF    85504      // 64 f32
#define ETIN_OFF  85760      // [256 e][264 f] bf16, stride 528 B
#define ETIN_STR  528
#define H_OFF     220928     // 8 waves x [32 m][136 k] bf16, stride 272 B
#define H_STR     272
#define SMEM_BYTES 290560

union FragU { u32x4 q[2]; v16bf v; };

// A-matrix 16x32 bf16 fragment (ISA 7.12.2): lane m = lane&15, half = lane>>4.
// elements 0-7  : k = kb*32 + half*8  + 0..7   (16 B)
// elements 8-15 : k = kb*32 + 16 + half*8 + 0..7 (16 B, +32 B after first chunk)
__device__ __forceinline__ v16bf load_a_frag(const char* rowBase, int kb, int half) {
  FragU u;
  const char* p = rowBase + kb * 64 + half * 16;
  u.q[0] = *(const u32x4*)(p);
  u.q[1] = *(const u32x4*)(p + 32);
  return u.v;
}

// B-matrix 32x16 bf16 fragment: lane n = lane&15; same k order as A:
// elements 0-7 k = kb*32 + half*8 + e, elements 8-15 k = kb*32 + 16 + half*8 + (e-8).
__device__ __forceinline__ v16bf load_b_frag(const char* rowBase, int kb, int half) {
  FragU u;
  const char* p = rowBase + kb * 64 + half * 16;
  u.q[0] = *(const u32x4*)(p);
  u.q[1] = *(const u32x4*)(p + 32);
  return u.v;
}

__global__ __launch_bounds__(256) void edgeblock_wmma_kernel(
    const float* __restrict__ vdata, const float* __restrict__ edata,
    const float* __restrict__ cdata, const float* __restrict__ W1,
    const float* __restrict__ b1,    const float* __restrict__ W2,
    const float* __restrict__ b2,    const int* __restrict__ sid,
    const int* __restrict__ rid,     float* __restrict__ out,
    int B, int N, int E, int numTiles, int tilesPerBatch)
{
  extern __shared__ char smem[];
  const int tid = threadIdx.x;

  // ---- Stage weights (fp32 -> bf16, transposed) + biases into LDS, once ----
  for (int idx = tid; idx < 256 * 128; idx += 256) {   // W1 [256 k][128 n]
    int k = idx >> 7, n = idx & 127;
    ((__bf16*)(smem + W1T_OFF + n * W1T_STR))[k] = (__bf16)W1[idx];
  }
  for (int idx = tid; idx < 128 * 64; idx += 256) {    // W2 [128 k][64 n]
    int k = idx >> 6, n = idx & 63;
    ((__bf16*)(smem + W2T_OFF + n * W2T_STR))[k] = (__bf16)W2[idx];
  }
  if (tid < 128) ((float*)(smem + B1_OFF))[tid] = b1[tid];
  if (tid < 64)  ((float*)(smem + B2_OFF))[tid] = b2[tid];
  __syncthreads();

  const int lane = tid & 31;
  const int wave = tid >> 5;
  const int n16  = lane & 15;
  const int half = lane >> 4;

  for (int tile = blockIdx.x; tile < numTiles; tile += gridDim.x) {
    const int bb    = tile / tilesPerBatch;
    const int e0    = (tile - bb * tilesPerBatch) << 8;   // 256 edges per tile
    const int valid = (E - e0) < 256 ? (E - e0) : 256;    // rows valid this tile
    const float* vb = vdata + (size_t)bb * N * 64;
    const float* eb = edata + ((size_t)bb * E + e0) * 64;
    const int*   sp = sid + (size_t)bb * E + e0;
    const int*   rp = rid + (size_t)bb * E + e0;
    const float* cb = cdata + bb * 64;

    // ---- Build etin tile [256][256] bf16 = [edge | sender | receiver | ctx] ----
    for (int idx = tid; idx < 256 * 64; idx += 256) {
      int e = idx >> 6, f = idx & 63;
      int ec = (e < valid) ? e : (valid - 1);             // clamp tail rows
      __bf16* row = (__bf16*)(smem + ETIN_OFF + e * ETIN_STR);
      float ev = eb[(size_t)ec * 64 + f];
      float sv = vb[(size_t)sp[ec] * 64 + f];
      float rv = vb[(size_t)rp[ec] * 64 + f];
      float cv = cb[f];
      row[f]       = (__bf16)ev;
      row[64 + f]  = (__bf16)sv;
      row[128 + f] = (__bf16)rv;
      row[192 + f] = (__bf16)cv;
    }
    __syncthreads();

    // ---- Prefetch next tile's edata while this tile computes ----
    {
      int nt = tile + gridDim.x;
      if (nt < numTiles) {
        int nbb = nt / tilesPerBatch;
        int ne0 = (nt - nbb * tilesPerBatch) << 8;
        if (ne0 + 256 <= E) {
          const float* neb = edata + ((size_t)nbb * E + ne0) * 64;
          __builtin_prefetch(neb + (size_t)tid * 64, 0, 0);  // 256 B/thread = 64 KB
        }
      }
    }

    // ---- GEMM1: [32 x 256] x [256 x 128], M register-blocked x2 ----
    v8f acc[2][8];
#pragma unroll
    for (int nb = 0; nb < 8; ++nb) {
      float bias = ((const float*)(smem + B1_OFF))[nb * 16 + n16];
#pragma unroll
      for (int r = 0; r < 8; ++r) { acc[0][nb][r] = bias; acc[1][nb][r] = bias; }
    }
    const int m0 = wave * 32;
    const char* arow0 = smem + ETIN_OFF + (m0 + n16) * ETIN_STR;
    const char* arow1 = arow0 + 16 * ETIN_STR;
#pragma unroll
    for (int kb = 0; kb < 8; ++kb) {
      v16bf a0 = load_a_frag(arow0, kb, half);
      v16bf a1 = load_a_frag(arow1, kb, half);
#pragma unroll
      for (int nb = 0; nb < 8; ++nb) {
        v16bf bm = load_b_frag(smem + W1T_OFF + (nb * 16 + n16) * W1T_STR, kb, half);
        acc[0][nb] = __builtin_amdgcn_wmma_f32_16x16x32_bf16(
            false, a0, false, bm, (short)0, acc[0][nb], false, false);
        acc[1][nb] = __builtin_amdgcn_wmma_f32_16x16x32_bf16(
            false, a1, false, bm, (short)0, acc[1][nb], false, false);
      }
    }

    // ---- ReLU, bf16, stash h tile [32][128] in per-wave LDS region ----
    char* hbase = smem + H_OFF + wave * 32 * H_STR;
#pragma unroll
    for (int mt = 0; mt < 2; ++mt) {
#pragma unroll
      for (int nb = 0; nb < 8; ++nb) {
#pragma unroll
        for (int r = 0; r < 8; ++r) {
          float v = acc[mt][nb][r];
          v = v > 0.0f ? v : 0.0f;
          ((__bf16*)(hbase + (mt * 16 + r + 8 * half) * H_STR))[nb * 16 + n16] = (__bf16)v;
        }
      }
    }
    // per-wave LDS RAW: same-wave DS ops are in-order; wait to be safe
    asm volatile("s_wait_dscnt 0" ::: "memory");

    // ---- GEMM2: [32 x 128] x [128 x 64] ----
    v8f acc2[2][4];
#pragma unroll
    for (int nb = 0; nb < 4; ++nb) {
      float bias = ((const float*)(smem + B2_OFF))[nb * 16 + n16];
#pragma unroll
      for (int r = 0; r < 8; ++r) { acc2[0][nb][r] = bias; acc2[1][nb][r] = bias; }
    }
    const char* h0 = hbase + n16 * H_STR;
    const char* h1 = hbase + (16 + n16) * H_STR;
#pragma unroll
    for (int kb = 0; kb < 4; ++kb) {
      v16bf a0 = load_a_frag(h0, kb, half);
      v16bf a1 = load_a_frag(h1, kb, half);
#pragma unroll
      for (int nb = 0; nb < 4; ++nb) {
        v16bf bm = load_b_frag(smem + W2T_OFF + (nb * 16 + n16) * W2T_STR, kb, half);
        acc2[0][nb] = __builtin_amdgcn_wmma_f32_16x16x32_bf16(
            false, a0, false, bm, (short)0, acc2[0][nb], false, false);
        acc2[1][nb] = __builtin_amdgcn_wmma_f32_16x16x32_bf16(
            false, a1, false, bm, (short)0, acc2[1][nb], false, false);
      }
    }

    // ---- f32 stores: out[b][e0 + wave*32 + row][n], tail rows predicated ----
    const int rowLimit = valid - m0;
    float* ob = out + ((size_t)bb * E + e0 + m0) * 64;
#pragma unroll
    for (int mt = 0; mt < 2; ++mt) {
#pragma unroll
      for (int nb = 0; nb < 4; ++nb) {
#pragma unroll
        for (int r = 0; r < 8; ++r) {
          int row = mt * 16 + r + 8 * half;
          if (row < rowLimit)
            ob[(size_t)row * 64 + nb * 16 + n16] = acc2[mt][nb][r];
        }
      }
    }
    __syncthreads();  // protect etin before next tile's fill
  }
}

extern "C" void kernel_launch(void* const* d_in, const int* in_sizes, int n_in,
                              void* d_out, int out_size, void* d_ws, size_t ws_size,
                              hipStream_t stream) {
  const float* vdata = (const float*)d_in[0];
  const float* edata = (const float*)d_in[1];
  const float* cdata = (const float*)d_in[2];
  const float* W1    = (const float*)d_in[3];
  const float* b1    = (const float*)d_in[4];
  const float* W2    = (const float*)d_in[5];
  const float* b2    = (const float*)d_in[6];
  const int*   sid   = (const int*)d_in[7];
  const int*   rid   = (const int*)d_in[8];
  float* out = (float*)d_out;

  const int B = in_sizes[2] / 64;            // cdata is [B,64]
  const int E = in_sizes[7] / B;             // sender_ids is [B,E]
  const int N = in_sizes[0] / (B * 64);      // vdata is [B,N,64]
  const int tilesPerBatch = (E + 255) / 256; // 256-edge tiles, tail predicated
  const int numTiles = B * tilesPerBatch;
  int grid = numTiles < 1280 ? numTiles : 1280;

  // Large dynamic-LDS launch (290,560 B <= 320 KB WGP budget)
  hipFuncSetAttribute((const void*)edgeblock_wmma_kernel,
                      hipFuncAttributeMaxDynamicSharedMemorySize, SMEM_BYTES);

  edgeblock_wmma_kernel<<<grid, 256, SMEM_BYTES, stream>>>(
      vdata, edata, cdata, W1, b1, W2, b2, sid, rid, out,
      B, N, E, numTiles, tilesPerBatch);
}
